// My_Glove_BLSTM_CRF_7395933684382
// MI455X (gfx1250) — matrix operand order repaired and checked by reading, not compiled, not run on hardware
//
#include <hip/hip_runtime.h>
#include <hip/hip_bf16.h>

typedef _Float16 v8h  __attribute__((ext_vector_type(8)));
typedef _Float16 v16h __attribute__((ext_vector_type(16)));
typedef float    v8f  __attribute__((ext_vector_type(8)));

#define Bc   128
#define Tc   512
#define DIN  102
#define Hc   256
#define G4   1024     // 4*H
#define DLIN 128
#define KT   9        // CRF tags
#define BT   65536    // B*T
#define DINP 128      // D_IN padded to K multiple of 32

// ---------------- WMMA fragment loaders (row-major f16 sources) ----------------
// A 16x32: lane l holds row (l&15); halves[0..15] = K kg..kg+15, kg=(l>=16)?16:0
__device__ __forceinline__ v16h load_a_frag(const _Float16* A, int lda, int lane, int kbase) {
    int m  = lane & 15;
    int kg = (lane >> 4) * 16;
    const _Float16* p = A + (size_t)m * lda + kbase + kg;
    union { v16h v; v8h h[2]; } u;
    u.h[0] = *(const v8h*)(p);
    u.h[1] = *(const v8h*)(p + 8);
    return u.v;
}
// B 32x16 from weights stored (N,K) row-major: lane l holds col (l&15); 16 contiguous K at kbase+((l>=16)?16:0)
__device__ __forceinline__ v16h load_b_frag(const _Float16* W, int ldb, int lane, int kbase) {
    int n  = lane & 15;
    int kg = (lane >> 4) * 16;
    return *(const v16h*)(W + (size_t)n * ldb + kbase + kg);
}

__device__ __forceinline__ v8f wmma_f16(v16h a, v16h b, v8f c) {
    return __builtin_amdgcn_wmma_f32_16x16x32_f16(false, a, false, b, (short)0, c, false, false);
}

__device__ __forceinline__ float sigf(float x)  { return 1.0f / (1.0f + __expf(-x)); }
__device__ __forceinline__ float eluf(float x)  { return x > 0.0f ? x : (__expf(x) - 1.0f); }

// ---------------- f32 -> f16 convert with (N,K) zero padding ----------------
__global__ void k_pad_f16(const float* __restrict__ src, _Float16* __restrict__ dst,
                          int N, int K, int Np, int Kp) {
    int i = blockIdx.x * blockDim.x + threadIdx.x;
    if (i >= Np * Kp) return;
    int n = i / Kp, k = i - n * Kp;
    dst[i] = (n < N && k < K) ? (_Float16)src[(size_t)n * K + k] : (_Float16)0.0f;
}

// ---------------- gx = x @ Wih^T + bias  (both directions) ----------------
// A rows are (b*T + t); output stored gx[dir][t*B + b][G4] (f16)
__global__ __launch_bounds__(256) void k_gemm_gx(const _Float16* __restrict__ x_h,
                                                 const _Float16* __restrict__ Wih_h,
                                                 const float* __restrict__ b_f,
                                                 const float* __restrict__ b_b,
                                                 _Float16* __restrict__ gx) {
    int lane = threadIdx.x & 31, w = threadIdx.x >> 5;
    int rt  = blockIdx.x;                 // 0..4095 row tiles
    int ct  = blockIdx.y * 8 + w;         // 0..63 col tiles
    int dir = blockIdx.z;
    const _Float16* A  = x_h + (size_t)rt * 16 * DINP;
    const _Float16* Bw = Wih_h + (size_t)dir * G4 * DINP + (size_t)ct * 16 * DINP;
    const float* bias = dir ? b_b : b_f;
    v8f acc = {};
#pragma unroll
    for (int kt = 0; kt < DINP / 32; ++kt) {
        v16h a = load_a_frag(A, DINP, lane, kt * 32);
        v16h b = load_b_frag(Bw, DINP, lane, kt * 32);
        acc = wmma_f16(a, b, acc);
    }
    int nn = lane & 15, mb = (lane >> 4) * 8;
    float bv = bias[ct * 16 + nn];
    _Float16* out = gx + (size_t)dir * ((size_t)BT * G4);
#pragma unroll
    for (int r = 0; r < 8; ++r) {
        int g = rt * 16 + mb + r;         // row = b*T + t
        int bb = g >> 9;                  // / T (T=512)
        int t  = g & (Tc - 1);
        out[((size_t)t * Bc + bb) * G4 + ct * 16 + nn] = (_Float16)(acc[r] + bv);
    }
}

// ---------------- recurrent LSTM: 16 blocks (8 batch-chunks x 2 dirs), 16 waves ----------------
__global__ __launch_bounds__(512) void k_lstm(const _Float16* __restrict__ Whh_h,
                                              const _Float16* __restrict__ gx,
                                              _Float16* __restrict__ hseq) {
    __shared__ __align__(32) _Float16 h_sh[16 * Hc];
    int dir  = blockIdx.x >> 3;
    int rc   = blockIdx.x & 7;            // batch chunk
    int lane = threadIdx.x & 31;
    int j    = threadIdx.x >> 5;          // wave = H column tile 0..15
    const _Float16* Wb = Whh_h + (size_t)dir * G4 * Hc;
    const _Float16* gxd = gx + (size_t)dir * ((size_t)BT * G4);

    for (int i = threadIdx.x; i < 16 * Hc; i += 512) h_sh[i] = (_Float16)0.0f;
    float c[8];
#pragma unroll
    for (int r = 0; r < 8; ++r) c[r] = 0.0f;
    int nn = lane & 15, mb = (lane >> 4) * 8;
    __syncthreads();

#pragma unroll 1
    for (int s = 0; s < Tc; ++s) {
        int t = dir ? (Tc - 1 - s) : s;
        v8f acc[4] = {};
#pragma unroll
        for (int kt = 0; kt < Hc / 32; ++kt) {
            v16h a = load_a_frag(h_sh, Hc, lane, kt * 32);   // reused across 4 gates
#pragma unroll
            for (int g = 0; g < 4; ++g) {
                v16h b = load_b_frag(Wb + (size_t)(g * 16 + j) * 16 * Hc, Hc, lane, kt * 32);
                acc[g] = wmma_f16(a, b, acc[g]);
            }
        }
        _Float16 hv[8];
#pragma unroll
        for (int r = 0; r < 8; ++r) {
            int m = mb + r;
            int bglob = rc * 16 + m;
            size_t base = ((size_t)t * Bc + bglob) * G4 + j * 16 + nn;
            float gi = acc[0][r] + (float)gxd[base + 0 * Hc];
            float gf = acc[1][r] + (float)gxd[base + 1 * Hc];
            float gg = acc[2][r] + (float)gxd[base + 2 * Hc];
            float go = acc[3][r] + (float)gxd[base + 3 * Hc];
            float cn = sigf(gf) * c[r] + sigf(gi) * tanhf(gg);
            float hn = sigf(go) * tanhf(cn);
            c[r] = cn;
            hv[r] = (_Float16)hn;
            hseq[((size_t)t * Bc + bglob) * (2 * Hc) + dir * Hc + j * 16 + nn] = hv[r];
        }
        if (s + 1 < Tc) {
            int tn = dir ? (Tc - 2 - s) : (s + 1);
            __builtin_prefetch(gxd + ((size_t)tn * Bc + rc * 16 + mb) * G4 + j * 16 + nn, 0, 1);
        }
        __syncthreads();   // all A-fragment reads of h_sh done before overwrite
#pragma unroll
        for (int r = 0; r < 8; ++r) h_sh[(mb + r) * Hc + j * 16 + nn] = hv[r];
        __syncthreads();   // new h visible to all waves
    }
}

// ---------------- feats = ELU(hseq @ Wlin^T + blin) ----------------
__global__ __launch_bounds__(256) void k_feats(const _Float16* __restrict__ hseq,
                                               const _Float16* __restrict__ Wlin_h,
                                               const float* __restrict__ blin,
                                               _Float16* __restrict__ feats) {
    int lane = threadIdx.x & 31, w = threadIdx.x >> 5;
    int rt = blockIdx.x;                  // 0..4095
    int ct = w;                           // 0..7 -> 128 cols
    const _Float16* A  = hseq + (size_t)rt * 16 * (2 * Hc);
    const _Float16* Bw = Wlin_h + (size_t)ct * 16 * (2 * Hc);
    v8f acc = {};
#pragma unroll
    for (int kt = 0; kt < (2 * Hc) / 32; ++kt) {
        v16h a = load_a_frag(A, 2 * Hc, lane, kt * 32);
        v16h b = load_b_frag(Bw, 2 * Hc, lane, kt * 32);
        acc = wmma_f16(a, b, acc);
    }
    int nn = lane & 15, mb = (lane >> 4) * 8;
    float bv = blin[ct * 16 + nn];
#pragma unroll
    for (int r = 0; r < 8; ++r)
        feats[(size_t)(rt * 16 + mb + r) * DLIN + ct * 16 + nn] = (_Float16)eluf(acc[r] + bv);
}

// ---------------- emissions = feats @ Wcls^T + bcls (N padded 9->16, f32 out) ----------------
__global__ __launch_bounds__(256) void k_emis(const _Float16* __restrict__ feats,
                                              const _Float16* __restrict__ Wcls_h,
                                              const float* __restrict__ bcls,
                                              float* __restrict__ em) {
    int lane = threadIdx.x & 31, w = threadIdx.x >> 5;
    int rt = blockIdx.x * 8 + w;          // 0..4095
    const _Float16* A = feats + (size_t)rt * 16 * DLIN;
    v8f acc = {};
#pragma unroll
    for (int kt = 0; kt < DLIN / 32; ++kt) {
        v16h a = load_a_frag(A, DLIN, lane, kt * 32);
        v16h b = load_b_frag(Wcls_h, DLIN, lane, kt * 32);
        acc = wmma_f16(a, b, acc);
    }
    int nn = lane & 15, mb = (lane >> 4) * 8;
    float bv = (nn < KT) ? bcls[nn] : 0.0f;
#pragma unroll
    for (int r = 0; r < 8; ++r)
        em[(size_t)(rt * 16 + mb + r) * 16 + nn] = acc[r] + bv;
}

// ---------------- CRF NLL (one block, 128 threads; em is [t*B+b][16]) ----------------
__device__ __forceinline__ float lse9(const float* v) {
    float m = v[0];
#pragma unroll
    for (int i = 1; i < KT; ++i) m = fmaxf(m, v[i]);
    float s = 0.0f;
#pragma unroll
    for (int i = 0; i < KT; ++i) s += __expf(v[i] - m);
    return m + __logf(s);
}

__global__ __launch_bounds__(128) void k_crf(const float* __restrict__ em,
                                             const int* __restrict__ labels,
                                             const unsigned char* __restrict__ mask,
                                             const float* __restrict__ start_t,
                                             const float* __restrict__ end_t,
                                             const float* __restrict__ trans,
                                             float* __restrict__ out) {
    __shared__ float tr[KT * KT], st[KT], et[KT], red[128];
    int b = threadIdx.x;
    if (b < KT * KT) tr[b] = trans[b];
    if (b < KT) { st[b] = start_t[b]; et[b] = end_t[b]; }
    __syncthreads();

    float alpha[KT];
    int l0 = labels[b * Tc + 0];
    float num = st[l0] + em[(size_t)(0 * Bc + b) * 16 + l0];
#pragma unroll
    for (int j = 0; j < KT; ++j) alpha[j] = st[j] + em[(size_t)(0 * Bc + b) * 16 + j];
    int prev = l0;
    int cnt = mask[b * Tc + 0] ? 1 : 0;

    for (int t = 1; t < Tc; ++t) {
        float m = mask[b * Tc + t] ? 1.0f : 0.0f;
        int lt = labels[b * Tc + t];
        const float* emt = em + (size_t)(t * Bc + b) * 16;
        num += m * (tr[prev * KT + lt] + emt[lt]);
        prev = lt;
        cnt += mask[b * Tc + t] ? 1 : 0;
        float na[KT];
#pragma unroll
        for (int j = 0; j < KT; ++j) {
            float v[KT];
#pragma unroll
            for (int i = 0; i < KT; ++i) v[i] = alpha[i] + tr[i * KT + j];
            na[j] = lse9(v) + emt[j];
        }
#pragma unroll
        for (int j = 0; j < KT; ++j) alpha[j] = (m > 0.0f) ? na[j] : alpha[j];
    }
    num += et[labels[b * Tc + (cnt - 1)]];
    float v[KT];
#pragma unroll
    for (int j = 0; j < KT; ++j) v[j] = alpha[j] + et[j];
    float logZ = lse9(v);
    red[b] = -(num - logZ);
    __syncthreads();
    for (int off = 64; off > 0; off >>= 1) {
        if (b < off) red[b] += red[b + off];
        __syncthreads();
    }
    if (b == 0) out[0] = red[0];
}

// ---------------- host side ----------------
extern "C" void kernel_launch(void* const* d_in, const int* in_sizes, int n_in,
                              void* d_out, int out_size, void* d_ws, size_t ws_size,
                              hipStream_t stream) {
    const float* x       = (const float*)d_in[0];
    const int*   labels  = (const int*)d_in[2];
    const unsigned char* mask = (const unsigned char*)d_in[3];
    const float* Wih_f   = (const float*)d_in[4];
    const float* Whh_f   = (const float*)d_in[5];
    const float* b_f     = (const float*)d_in[6];
    const float* Wih_b   = (const float*)d_in[7];
    const float* Whh_b   = (const float*)d_in[8];
    const float* b_b     = (const float*)d_in[9];
    const float* Wlin    = (const float*)d_in[10];
    const float* blin    = (const float*)d_in[11];
    const float* Wcls    = (const float*)d_in[12];
    const float* bcls    = (const float*)d_in[13];
    const float* start_t = (const float*)d_in[14];
    const float* end_t   = (const float*)d_in[15];
    const float* trans   = (const float*)d_in[16];

    char* ws = (char*)d_ws;
    size_t off = 0;
    auto carve = [&](size_t bytes) { size_t o = off; off += (bytes + 255) & ~(size_t)255; return o; };
    _Float16* x_h    = (_Float16*)(ws + carve((size_t)BT * DINP * 2));
    _Float16* Wih_h  = (_Float16*)(ws + carve((size_t)2 * G4 * DINP * 2));
    _Float16* Whh_h  = (_Float16*)(ws + carve((size_t)2 * G4 * Hc * 2));
    _Float16* Wlin_h = (_Float16*)(ws + carve((size_t)DLIN * 2 * Hc * 2));
    _Float16* Wcls_h = (_Float16*)(ws + carve((size_t)16 * DLIN * 2));
    _Float16* gx     = (_Float16*)(ws + carve((size_t)2 * BT * G4 * 2));
    _Float16* hseq   = (_Float16*)(ws + carve((size_t)BT * 2 * Hc * 2));
    _Float16* feats  = (_Float16*)(ws + carve((size_t)BT * DLIN * 2));
    float*    em     = (float*)   (ws + carve((size_t)BT * 16 * 4));

    auto pad = [&](const float* src, _Float16* dst, int N, int K, int Np, int Kp) {
        int total = Np * Kp;
        k_pad_f16<<<(total + 255) / 256, 256, 0, stream>>>(src, dst, N, K, Np, Kp);
    };
    pad(x,     x_h,                 BT,   DIN,  BT,   DINP);
    pad(Wih_f, Wih_h,               G4,   DIN,  G4,   DINP);
    pad(Wih_b, Wih_h + G4 * DINP,   G4,   DIN,  G4,   DINP);
    pad(Whh_f, Whh_h,               G4,   Hc,   G4,   Hc);
    pad(Whh_b, Whh_h + G4 * Hc,     G4,   Hc,   G4,   Hc);
    pad(Wlin,  Wlin_h,              DLIN, 2*Hc, DLIN, 2*Hc);
    pad(Wcls,  Wcls_h,              KT,   DLIN, 16,   DLIN);

    k_gemm_gx<<<dim3(BT / 16, 8, 2), 256, 0, stream>>>(x_h, Wih_h, b_f, b_b, gx);
    k_lstm<<<16, 512, 0, stream>>>(Whh_h, gx, hseq);
    k_feats<<<BT / 16, 256, 0, stream>>>(hseq, Wlin_h, blin, feats);
    k_emis<<<BT / 128, 256, 0, stream>>>(feats, Wcls_h, bcls, em);
    k_crf<<<1, 128, 0, stream>>>(em, labels, mask, start_t, end_t, trans, (float*)d_out);
}
